// WeatherPrediction_54073638257190
// MI455X (gfx1250) — matrix operand order repaired and checked by reading, not compiled, not run on hardware
//
#include <hip/hip_runtime.h>

// ---------------------------------------------------------------------------
// GraphCast-style GNN for MI455X (gfx1250, wave32, WMMA f16 -> f32 acc)
// ---------------------------------------------------------------------------
#define N_SPATIAL 16380           // 91*180
#define N_SPHERE  2883
#define N_TOTAL   19263
#define N_EDGES   65520           // 4*N_SPATIAL
#define NFEAT     78
#define LATENT    256

typedef __attribute__((ext_vector_type(16))) _Float16 v16h;
typedef __attribute__((ext_vector_type(8)))  _Float16 v8h;
typedef __attribute__((ext_vector_type(4)))  _Float16 v4h;
typedef __attribute__((ext_vector_type(8)))  float    v8f;

__device__ __forceinline__ v16h cat8(v8h lo, v8h hi) {
  return __builtin_shufflevector(lo, hi, 0,1,2,3,4,5,6,7,8,9,10,11,12,13,14,15);
}

// --------------------------- helper kernels --------------------------------

// Transpose+convert W[k][n] (fp32, lda=N) -> WT[n][koff+k] (f16, lda=Kpad),
// zero-padding k>=Kreal and n>=N.  Writes region [0,256) x [koff,koff+Kthis).
__global__ void fill_wt_kernel(const float* __restrict__ W, _Float16* __restrict__ WT,
                               int Kreal, int N, int Kpad, int koff, int Kthis) {
  int id = blockIdx.x * 256 + threadIdx.x;
  int tot = 256 * Kthis;
  if (id >= tot) return;
  int n = id / Kthis, k = id % Kthis;
  float v = (k < Kreal && n < N) ? W[(size_t)k * N + n] : 0.f;
  WT[(size_t)n * Kpad + koff + k] = (_Float16)v;
}

__global__ void zero_f32_kernel(float* __restrict__ p, long n) {
  long i = (long)blockIdx.x * blockDim.x + threadIdx.x;
  long st = (long)gridDim.x * blockDim.x;
  for (; i < n; i += st) p[i] = 0.f;
}

__global__ void copy_f32_kernel(float* __restrict__ dst, const float* __restrict__ src, long n) {
  long i = (long)blockIdx.x * blockDim.x + threadIdx.x;
  long st = (long)gridDim.x * blockDim.x;
  for (; i < n; i += st) dst[i] = src[i];
}

// dst[r][c] = row[c]
__global__ void bcast_row_kernel(float* __restrict__ dst, const float* __restrict__ row, long n) {
  long i = (long)blockIdx.x * blockDim.x + threadIdx.x;
  long st = (long)gridDim.x * blockDim.x;
  for (; i < n; i += st) dst[i] = row[i & 255];
}

// agg[rcv[e]][c] += ev[e][c]   (fp32 global atomics -> L2 atomic units)
__global__ void seg_scatter_kernel(const float* __restrict__ ev, const int* __restrict__ rcv,
                                   float* __restrict__ agg, int ne) {
  int e = blockIdx.x;
  int c = threadIdx.x;
  int r = rcv[e];
  atomicAdd(&agg[(size_t)r * 256 + c], ev[(size_t)e * 256 + c]);
}

// nm[r][c] = mean_j h[nbr[r][j]][c]  over 6 neighbors
__global__ void neigh_mean_kernel(const float* __restrict__ h, const int* __restrict__ nbr,
                                  float* __restrict__ nm) {
  int r = blockIdx.x;
  int c = threadIdx.x;
  const int* nb = nbr + (size_t)r * 6;
  float s = 0.f;
#pragma unroll
  for (int j = 0; j < 6; ++j) s += h[(size_t)nb[j] * 256 + c];
  nm[(size_t)r * 256 + c] = s * (1.f / 6.f);
}

// ---------------------------------------------------------------------------
// Fused layer:  out = [relu_after](LN([relu_before](A @ W + bias [+ skip])))
// NSEG==1 : A = plain [M x Kreal] (zero-padded to Kpad); VEC4 iff Kreal%4==0.
// NSEG>=2 : A rows = concat of NSEG 256-wide segments, seg s gathered via
//           index array (or identity), K = NSEG*256.
// Block = 256 threads (8 waves); tile = 32 rows x 256 cols.
// Whole 256-wide K-slab staged in LDS -> 8 K-steps x 4 WMMA per barrier pair.
// ---------------------------------------------------------------------------
template <int NSEG, bool VEC4>
__global__ void __launch_bounds__(256)
mlp_layer_kernel(const float* __restrict__ a0, const float* __restrict__ a1,
                 const float* __restrict__ a2,
                 const int* __restrict__ i1, const int* __restrict__ i2,
                 int M, int Kreal, int Kpad,
                 const _Float16* __restrict__ WT,          // [256][Kpad] f16
                 const float* __restrict__ bias,
                 const float* __restrict__ gamma,
                 const float* __restrict__ betav,
                 const float* __restrict__ skip,           // [M][256] or null
                 float* __restrict__ out, int n_out,
                 int relu_before, int relu_after) {
  // A-slab (32 x 264 f16, 16.9KB) aliases the LN buffer (32 x 264 f32, 33.8KB):
  // all A reads complete (barrier) before Ys is written.
  __shared__ __align__(16) unsigned char smem[32 * 264 * 4];
  _Float16* As = (_Float16*)smem;      // stride 264 halfs
  float*    Ys = (float*)smem;         // stride 264 floats

  const int tid  = threadIdx.x;
  const int wave = tid >> 5;
  const int lane = tid & 31;
  const int r0   = blockIdx.x * 32;
  const int hi   = lane >> 4;          // lane half (0/1)
  const int lan  = lane & 15;
  const int n0   = wave * 32 + lan;    // output column, B-tile 0
  const int n1   = n0 + 16;            // output column, B-tile 1

  v8f z8 = {0.f, 0.f, 0.f, 0.f, 0.f, 0.f, 0.f, 0.f};
  v8f acc00 = z8, acc01 = z8, acc10 = z8, acc11 = z8;

  const int nslab = (NSEG >= 2) ? NSEG : ((Kpad + 255) >> 8);
  for (int sl = 0; sl < nslab; ++sl) {
    const int s0 = sl << 8;
    const int wslab = (NSEG >= 2) ? 256 : min(256, Kpad - s0);

    // ---- stage A slab (32 rows x wslab) into LDS as f16 ----
    if (VEC4) {
      const int nc4 = wslab >> 2;
      for (int e = tid; e < 32 * nc4; e += 256) {
        int row  = e / nc4;
        int c4   = (e - row * nc4) << 2;
        int grow = r0 + row;
        float4 v = make_float4(0.f, 0.f, 0.f, 0.f);
        if (grow < M) {
          const float* src;
          if (NSEG == 1) {
            src = a0 + (size_t)grow * Kreal + s0 + c4;
          } else {
            const float* p  = (sl == 0) ? a0 : ((sl == 1) ? a1 : a2);
            const int*   ix = (sl == 0) ? (const int*)nullptr : ((sl == 1) ? i1 : i2);
            int rr = ix ? ix[grow] : grow;
            src = p + (size_t)rr * 256 + c4;
          }
          v = *(const float4*)src;
        }
        v4h h4 = {(_Float16)v.x, (_Float16)v.y, (_Float16)v.z, (_Float16)v.w};
        *(v4h*)&As[row * 264 + c4] = h4;
      }
    } else {
      for (int e = tid; e < 32 * wslab; e += 256) {
        int row  = e / wslab;
        int col  = e - row * wslab;
        int grow = r0 + row;
        int kc   = s0 + col;
        float v = (grow < M && kc < Kreal) ? a0[(size_t)grow * Kreal + kc] : 0.f;
        As[row * 264 + col] = (_Float16)v;
      }
    }
    __syncthreads();

    // ---- K-steps over the slab: 4 WMMA each, no barriers in between ----
    const int ksteps = wslab >> 5;
#pragma unroll 8
    for (int ks = 0; ks < ksteps; ++ks) {
      const int kin = ks << 5;
      // A operands (16-bit 16x32 layout: lanes 0-15 K 0-7/16-23, lanes 16-31 +8)
      const _Float16* ap0 = &As[lan * 264 + kin + (hi << 3)];
      const _Float16* ap1 = ap0 + 16 * 264;
      v16h A0 = cat8(*(const v8h*)ap0, *(const v8h*)(ap0 + 16));
      v16h A1 = cat8(*(const v8h*)ap1, *(const v8h*)(ap1 + 16));
      // B operands (16-bit 32x16 layout: lanes 0-15 K 0-7/16-23, lanes 16-31 +8)
      const int kg = s0 + kin + (hi << 3);
      const _Float16* bp0 = WT + (size_t)n0 * Kpad + kg;
      const _Float16* bp1 = WT + (size_t)n1 * Kpad + kg;
      v16h B0 = cat8(*(const v8h*)bp0, *(const v8h*)(bp0 + 16));
      v16h B1 = cat8(*(const v8h*)bp1, *(const v8h*)(bp1 + 16));
      __builtin_prefetch((const void*)(bp0 + 32), 0, 1);   // next K slice (L2)
      __builtin_prefetch((const void*)(bp1 + 32), 0, 1);

      acc00 = __builtin_amdgcn_wmma_f32_16x16x32_f16(false, A0, false, B0, (short)0, acc00, false, false);
      acc01 = __builtin_amdgcn_wmma_f32_16x16x32_f16(false, A0, false, B1, (short)0, acc01, false, false);
      acc10 = __builtin_amdgcn_wmma_f32_16x16x32_f16(false, A1, false, B0, (short)0, acc10, false, false);
      acc11 = __builtin_amdgcn_wmma_f32_16x16x32_f16(false, A1, false, B1, (short)0, acc11, false, false);
    }
    __syncthreads();   // all A reads done before restage / Ys aliasing
  }

  // ---- epilogue: bias (+skip) (+relu) -> LDS ----
  {
    float b0 = (n0 < n_out) ? bias[n0] : 0.f;
    float b1 = (n1 < n_out) ? bias[n1] : 0.f;
#pragma unroll
    for (int mt = 0; mt < 2; ++mt) {
      v8f a0v = mt ? acc10 : acc00;
      v8f a1v = mt ? acc11 : acc01;
#pragma unroll
      for (int v = 0; v < 8; ++v) {
        int row  = mt * 16 + (hi << 3) + v;   // C layout: VGPR v -> M=v (+8 hi)
        int grow = r0 + row;
        float y0 = a0v[v] + b0;
        float y1 = a1v[v] + b1;
        if (skip && grow < M) {
          y0 += skip[(size_t)grow * 256 + n0];
          y1 += skip[(size_t)grow * 256 + n1];
        }
        if (relu_before) { y0 = fmaxf(y0, 0.f); y1 = fmaxf(y1, 0.f); }
        Ys[row * 264 + n0] = y0;
        Ys[row * 264 + n1] = y1;
      }
    }
  }
  __syncthreads();

  // ---- LayerNorm over n_out columns; wave w handles rows 4w..4w+3 ----
  const float inv_n = 1.f / (float)n_out;
#pragma unroll
  for (int rr = 0; rr < 4; ++rr) {
    int row  = wave * 4 + rr;
    int grow = r0 + row;
    if (grow >= M) continue;
    float yv[8];
    float s1 = 0.f, s2 = 0.f;
#pragma unroll
    for (int iC = 0; iC < 8; ++iC) {
      int c = lane + 32 * iC;
      float y = Ys[row * 264 + c];
      yv[iC] = y;
      if (c < n_out) { s1 += y; s2 += y * y; }
    }
#pragma unroll
    for (int m = 16; m >= 1; m >>= 1) {   // wave32 reduction
      s1 += __shfl_xor(s1, m, 32);
      s2 += __shfl_xor(s2, m, 32);
    }
    float mu  = s1 * inv_n;
    float var = s2 * inv_n - mu * mu;
    float rs  = rsqrtf(var + 1e-5f);
#pragma unroll
    for (int iC = 0; iC < 8; ++iC) {
      int c = lane + 32 * iC;
      if (c < n_out) {
        float o = (yv[iC] - mu) * rs * gamma[c] + betav[c];
        if (relu_after) o = fmaxf(o, 0.f);
        out[(size_t)grow * n_out + c] = o;
      }
    }
  }
}

// ---------------------------------------------------------------------------
// Host orchestration.  Input order = setup_inputs() dict insertion order,
// params flattened recursively in insertion order (171 leaves total):
//   enc:  sender[W,b,g,beta]x2 =0..7, receiver=8..15, edge0=16..23,
//         edge_mp[3]=24..47, node_mp[3]=48..71
//   dec:  sender=72..79, receiver=80..87, edge0=88..95,
//         edge_mp[3]=96..119, node_mp[3]=120..143, final=144..151
//   proc: [w_self,w_neigh,b_neigh,g,beta] x2 = 152..161
//   latlon=162 sphere_init=163 enc_edge=164 dec_edge=165
//   enc_snd=166 enc_rcv=167 dec_snd=168 dec_rcv=169 neighbor_idx=170
// ---------------------------------------------------------------------------
extern "C" void kernel_launch(void* const* d_in, const int* in_sizes, int n_in,
                              void* d_out, int out_size, void* d_ws, size_t ws_size,
                              hipStream_t stream) {
  (void)in_sizes; (void)n_in; (void)out_size; (void)ws_size;

  auto F = [&](int i) { return (const float*)d_in[i]; };
  auto I = [&](int i) { return (const int*)d_in[i]; };

  // ---- carve workspace ----
  char* wsb = (char*)d_ws;
  size_t off = 0;
  auto alloc_f = [&](size_t n) -> float* { float* p = (float*)(wsb + off); off += n * sizeof(float); return p; };
  float* nodes = alloc_f((size_t)N_TOTAL * 256);
  float* edges = alloc_f((size_t)N_EDGES * 256);
  float* agg   = alloc_f((size_t)N_TOTAL * 256);
  float* tmp   = alloc_f((size_t)N_EDGES * 256);
  float* h1    = alloc_f((size_t)N_SPHERE * 256);
  float* h2    = alloc_f((size_t)N_SPHERE * 256);
  float* nm    = alloc_f((size_t)N_SPHERE * 256);
  float* zbuf  = alloc_f(32 * 256);
  float* crow1 = alloc_f(32 * 256);
  float* crow2 = alloc_f(32 * 256);
  off = (off + 255) & ~(size_t)255;
  _Float16* wt = (_Float16*)(wsb + off);
  size_t wcur = 0;

  auto fillW = [&](int widx, int Kreal, int N, int Kpad) -> const _Float16* {
    _Float16* dst = wt + wcur;
    wcur += (size_t)256 * Kpad;
    int tot = 256 * Kpad;
    fill_wt_kernel<<<(tot + 255) / 256, 256, 0, stream>>>(F(widx), dst, Kreal, N, Kpad, 0, Kpad);
    return dst;
  };

  // ---- pre-transpose all weights to f16 [N=256][Kpad] ----
  const _Float16* W_es0 = fillW(0, NFEAT, 256, 96);
  const _Float16* W_es1 = fillW(4, 256, 256, 256);
  const _Float16* W_er0 = fillW(8, 256, 256, 256);
  const _Float16* W_er1 = fillW(12, 256, 256, 256);
  const _Float16* W_ee0 = fillW(16, 3, 256, 32);
  const _Float16* W_ee1 = fillW(20, 256, 256, 256);
  const _Float16* W_eemp0[3]; const _Float16* W_eemp1[3];
  const _Float16* W_enmp0[3]; const _Float16* W_enmp1[3];
  for (int s = 0; s < 3; ++s) {
    W_eemp0[s] = fillW(24 + 8 * s, 768, 256, 768);
    W_eemp1[s] = fillW(28 + 8 * s, 256, 256, 256);
    W_enmp0[s] = fillW(48 + 8 * s, 512, 256, 512);
    W_enmp1[s] = fillW(52 + 8 * s, 256, 256, 256);
  }
  const _Float16* W_ds0 = fillW(72, 256, 256, 256);
  const _Float16* W_ds1 = fillW(76, 256, 256, 256);
  const _Float16* W_dr0 = fillW(80, 256, 256, 256);
  const _Float16* W_dr1 = fillW(84, 256, 256, 256);
  const _Float16* W_de0 = fillW(88, 3, 256, 32);
  const _Float16* W_de1 = fillW(92, 256, 256, 256);
  const _Float16* W_demp0[3]; const _Float16* W_demp1[3];
  const _Float16* W_dnmp0[3]; const _Float16* W_dnmp1[3];
  for (int s = 0; s < 3; ++s) {
    W_demp0[s] = fillW(96 + 8 * s, 768, 256, 768);
    W_demp1[s] = fillW(100 + 8 * s, 256, 256, 256);
    W_dnmp0[s] = fillW(120 + 8 * s, 512, 256, 512);
    W_dnmp1[s] = fillW(124 + 8 * s, 256, 256, 256);
  }
  const _Float16* W_fin0 = fillW(144, 256, 256, 256);
  const _Float16* W_fin1 = fillW(148, 256, NFEAT, 256);   // rows >=78 zeroed
  const _Float16* W_proc[2];
  for (int l = 0; l < 2; ++l) {            // combined [w_self ; w_neigh] -> K=512
    _Float16* dst = wt + wcur;
    wcur += (size_t)256 * 512;
    int tot = 256 * 256;
    fill_wt_kernel<<<(tot + 255) / 256, 256, 0, stream>>>(F(152 + 5 * l + 0), dst, 256, 256, 512, 0, 256);
    fill_wt_kernel<<<(tot + 255) / 256, 256, 0, stream>>>(F(152 + 5 * l + 1), dst, 256, 256, 512, 256, 256);
    W_proc[l] = dst;
  }

  auto layer = [&](const float* x0, const float* x1, const float* x2,
                   const int* ix1, const int* ix2, int nseg,
                   int M, int Kreal, int Kpad, const _Float16* WT,
                   const float* bias, const float* g, const float* be,
                   const float* skip, float* outp, int n_out, int rb, int ra) {
    dim3 grid((M + 31) / 32);
    if (nseg == 3)
      mlp_layer_kernel<3, true><<<grid, 256, 0, stream>>>(x0, x1, x2, ix1, ix2, M, Kreal, Kpad,
                                                          WT, bias, g, be, skip, outp, n_out, rb, ra);
    else if (nseg == 2)
      mlp_layer_kernel<2, true><<<grid, 256, 0, stream>>>(x0, x1, x2, ix1, ix2, M, Kreal, Kpad,
                                                          WT, bias, g, be, skip, outp, n_out, rb, ra);
    else if ((Kreal & 3) == 0)
      mlp_layer_kernel<1, true><<<grid, 256, 0, stream>>>(x0, x1, x2, ix1, ix2, M, Kreal, Kpad,
                                                          WT, bias, g, be, skip, outp, n_out, rb, ra);
    else
      mlp_layer_kernel<1, false><<<grid, 256, 0, stream>>>(x0, x1, x2, ix1, ix2, M, Kreal, Kpad,
                                                           WT, bias, g, be, skip, outp, n_out, rb, ra);
  };
  // 2-layer MLP (relu-then-LN at each layer), p = index of layer-0 W leaf
  auto mlp2 = [&](const float* X, int M, int K0real, int K0pad,
                  const _Float16* W0, int p0, const _Float16* W1, int p1,
                  float* dst, int n_out_last) {
    layer(X, nullptr, nullptr, nullptr, nullptr, 1, M, K0real, K0pad, W0,
          F(p0 + 1), F(p0 + 2), F(p0 + 3), nullptr, tmp, 256, 1, 0);
    layer(tmp, nullptr, nullptr, nullptr, nullptr, 1, M, 256, 256, W1,
          F(p1 + 1), F(p1 + 2), F(p1 + 3), nullptr, dst, n_out_last, 1, 0);
  };
  auto mp3 = [&](const int* snd, const int* rcv,
                 const _Float16* const* We0, const _Float16* const* We1, int eb,
                 const _Float16* const* Wn0, const _Float16* const* Wn1, int nb) {
    for (int s = 0; s < 3; ++s) {
      int e0 = eb + 8 * s, e1 = e0 + 4;
      // edges = MLP(concat[edges, nodes[snd], nodes[rcv]])
      layer(edges, nodes, nodes, snd, rcv, 3, N_EDGES, 768, 768, We0[s],
            F(e0 + 1), F(e0 + 2), F(e0 + 3), nullptr, tmp, 256, 1, 0);
      layer(tmp, nullptr, nullptr, nullptr, nullptr, 1, N_EDGES, 256, 256, We1[s],
            F(e1 + 1), F(e1 + 2), F(e1 + 3), nullptr, edges, 256, 1, 0);
      // agg = segment_sum(edges, rcv)
      zero_f32_kernel<<<2048, 256, 0, stream>>>(agg, (long)N_TOTAL * 256);
      seg_scatter_kernel<<<N_EDGES, 256, 0, stream>>>(edges, rcv, agg, N_EDGES);
      // nodes = MLP(concat[nodes, agg])
      int n0i = nb + 8 * s, n1i = n0i + 4;
      layer(nodes, agg, nullptr, nullptr, nullptr, 2, N_TOTAL, 512, 512, Wn0[s],
            F(n0i + 1), F(n0i + 2), F(n0i + 3), nullptr, tmp, 256, 1, 0);
      layer(tmp, nullptr, nullptr, nullptr, nullptr, 1, N_TOTAL, 256, 256, Wn1[s],
            F(n1i + 1), F(n1i + 2), F(n1i + 3), nullptr, nodes, 256, 1, 0);
    }
  };

  // ================= encoder =================
  mlp2(F(162), N_SPATIAL, NFEAT, 96, W_es0, 0, W_es1, 4, nodes, 256);
  mlp2(F(163), N_SPHERE, 256, 256, W_er0, 8, W_er1, 12, nodes + (size_t)N_SPATIAL * 256, 256);
  mlp2(F(164), N_EDGES, 3, 32, W_ee0, 16, W_ee1, 20, edges, 256);
  mp3(I(166), I(167), W_eemp0, W_eemp1, 24, W_enmp0, W_enmp1, 48);

  // ================= processor (GraphConv x2) =================
  copy_f32_kernel<<<2048, 256, 0, stream>>>(h1, nodes + (size_t)N_SPATIAL * 256, (long)N_SPHERE * 256);
  float* hc = h1;
  float* hn = h2;
  for (int l = 0; l < 2; ++l) {
    int pb = 152 + 5 * l;
    neigh_mean_kernel<<<N_SPHERE, 256, 0, stream>>>(hc, I(170), nm);
    // h = relu(LN(h + h@Wself + nm@Wneigh + b))  -> skip=h, relu AFTER LN
    layer(hc, nm, nullptr, nullptr, nullptr, 2, N_SPHERE, 512, 512, W_proc[l],
          F(pb + 2), F(pb + 3), F(pb + 4), hc, hn, 256, 0, 1);
    float* t = hc; hc = hn; hn = t;
  }

  // ================= decoder =================
  // recv0 = MLP(zeros) -> constant row, broadcast over spatial nodes
  zero_f32_kernel<<<32, 256, 0, stream>>>(zbuf, 32 * 256);
  layer(zbuf, nullptr, nullptr, nullptr, nullptr, 1, 16, 256, 256, W_dr0,
        F(81), F(82), F(83), nullptr, crow1, 256, 1, 0);
  layer(crow1, nullptr, nullptr, nullptr, nullptr, 1, 16, 256, 256, W_dr1,
        F(85), F(86), F(87), nullptr, crow2, 256, 1, 0);
  bcast_row_kernel<<<4096, 256, 0, stream>>>(nodes, crow2, (long)N_SPATIAL * 256);
  // send0 = MLP(h) into sphere rows of dnodes
  mlp2(hc, N_SPHERE, 256, 256, W_ds0, 72, W_ds1, 76, nodes + (size_t)N_SPATIAL * 256, 256);
  // dedges
  mlp2(F(165), N_EDGES, 3, 32, W_de0, 88, W_de1, 92, edges, 256);
  mp3(I(168), I(169), W_demp0, W_demp1, 96, W_dnmp0, W_dnmp1, 120);

  // ================= final MLP -> d_out (16380 x 78 fp32) =================
  layer(nodes, nullptr, nullptr, nullptr, nullptr, 1, N_SPATIAL, 256, 256, W_fin0,
        F(145), F(146), F(147), nullptr, tmp, 256, 1, 0);
  layer(tmp, nullptr, nullptr, nullptr, nullptr, 1, N_SPATIAL, 256, 256, W_fin1,
        F(149), F(150), F(151), nullptr, (float*)d_out, NFEAT, 1, 0);
}
